// TemporalExtLSTM_21792664060276
// MI455X (gfx1250) — matrix operand order repaired and checked by reading, not compiled, not run on hardware
//
#include <hip/hip_runtime.h>
#include <hip/hip_bf16.h>

// CDNA5 (gfx1250) fused graph->sequence->LSTM kernel.
// Only the first 18 edges of each graph matter (scatter mode='drop' with T=18),
// so we binary-search the sorted `batch` array for per-graph edge spans, then
// run the whole LSTM with one wave per 16 graphs using v_wmma_f32_16x16x32_f16.
// Gate activations use the gfx1250 hardware V_TANH_F32 transcendental
// (sigmoid(x) = 0.5*tanh(0.5x)+0.5) to avoid libm's branchy exp expansions.

typedef __attribute__((ext_vector_type(16))) _Float16 v16h;
typedef __attribute__((ext_vector_type(8)))  float    v8f;

#define T_HIDDEN 16
#define T_FEAT   18   // 11 node feats + 7 one-hot; also the truncated seq length
#define T_INS    11

union V16H { v16h v; _Float16 h[16]; };

// gfx1250 hardware tanh (TRANS op). Trailing v_nop satisfies the
// "1 independent op after TRANS before result use" hazard rule.
__device__ __forceinline__ float fast_tanh(float x) {
  float r;
  asm volatile("v_tanh_f32 %0, %1\n\tv_nop" : "=v"(r) : "v"(x));
  return r;
}
__device__ __forceinline__ float fast_sigmoid(float x) {
  return 0.5f * fast_tanh(0.5f * x) + 0.5f;
}

// first[g] = lower_bound(batch, g) for g in [0, ng); first[ng] = E.
__global__ void first_edge_kernel(const int* __restrict__ batch, int E, int ng,
                                  int* __restrict__ first) {
  int g = blockIdx.x * blockDim.x + threadIdx.x;
  if (g > ng) return;
  if (g == ng) { first[g] = E; return; }
  int lo = 0, hi = E;
  while (lo < hi) {
    int mid = (lo + hi) >> 1;
    if (batch[mid] < g) lo = mid + 1; else hi = mid;
  }
  first[g] = lo;
}

// One wave (32 lanes) handles a tile of 16 graphs. 8 waves / block -> 128 graphs.
__global__ __launch_bounds__(256) void lstm_fused_kernel(
    const float* __restrict__ x,     const float* __restrict__ W_ih,
    const float* __restrict__ W_hh,  const float* __restrict__ b_ih,
    const float* __restrict__ b_hh,  const float* __restrict__ W_fc,
    const float* __restrict__ b_fc,  const int* __restrict__ edge_index,
    const int* __restrict__ edge_attr, const int* __restrict__ first,
    float* __restrict__ out, int E) {
  __shared__ float lds_h[8][16][16];   // [wave][graph-row M][hidden N]

  const int lane  = threadIdx.x & 31;
  const int wave  = threadIdx.x >> 5;
  const int m     = lane & 15;         // row within the 16-graph tile
  const int hi    = lane >> 4;         // which half-wave (selects K slice)
  const int gbase = blockIdx.x * 128 + wave * 16;
  const int g     = gbase + m;

  // Per-graph edge span, truncated to first T_FEAT edges (faithful to reference).
  const int f0 = first[g];
  int cnt = first[g + 1] - f0;
  if (cnt > T_FEAT) cnt = T_FEAT;

  // ---- Preload weights into WMMA B-matrix layout (32x16 f16 per tile) ----
  // B element e in lane L: N = L&15, K = (e<8 ? e : e+8) + (L>=16 ? 8 : 0).
  V16H Bih[4], Bhh[4];
  float biasv[4];
  const int koff = hi * 8;
#pragma unroll
  for (int nt = 0; nt < 4; ++nt) {
    const int r = nt * 16 + m;         // gate row (torch order: i,f,g,o tiles)
#pragma unroll
    for (int e = 0; e < 16; ++e) {
      const int K = (e < 8) ? koff + e : 16 + koff + (e - 8);
      Bih[nt].h[e] = (K < T_FEAT)   ? (_Float16)W_ih[r * T_FEAT   + K] : (_Float16)0.0f;
      Bhh[nt].h[e] = (K < T_HIDDEN) ? (_Float16)W_hh[r * T_HIDDEN + K] : (_Float16)0.0f;
    }
    biasv[nt] = b_ih[r] + b_hh[r];
  }

  // ---- init h (LDS, D-layout mirror) and c (registers, D-layout) ----
#pragma unroll
  for (int v = 0; v < 8; ++v) lds_h[wave][v + hi * 8][m] = 0.0f;
  v8f cst;
#pragma unroll
  for (int v = 0; v < 8; ++v) cst[v] = 0.0f;
  __syncthreads();

  for (int t = 0; t < T_FEAT; ++t) {
    // ---- per-lane feature vector for graph row m at timestep t ----
    float fv[T_FEAT];
#pragma unroll
    for (int k = 0; k < T_FEAT; ++k) fv[k] = 0.0f;
    if (t < cnt) {
      const int j  = f0 + t;
      const int n1 = edge_index[j];
      const int n2 = edge_index[E + j];
      const int a  = edge_attr[j];
#pragma unroll
      for (int k = 0; k < T_INS; ++k) fv[k] = x[n1 * T_INS + k] + x[n2 * T_INS + k];
#pragma unroll
      for (int k = 0; k < 7; ++k) fv[T_INS + k] = (a == k) ? 1.0f : 0.0f;
    }

    // ---- pack A matrices (16x32 f16). Lane<16: K 0-7 / 16-23; lane>=16: K 8-15 / 24-31.
    V16H Ax, Ah;
    if (hi == 0) {
#pragma unroll
      for (int e = 0; e < 8; ++e) Ax.h[e] = (_Float16)fv[e];
      Ax.h[8] = (_Float16)fv[16];
      Ax.h[9] = (_Float16)fv[17];
#pragma unroll
      for (int e = 10; e < 16; ++e) Ax.h[e] = (_Float16)0.0f;
    } else {
#pragma unroll
      for (int e = 0; e < 8; ++e) Ax.h[e] = (_Float16)fv[8 + e];
#pragma unroll
      for (int e = 8; e < 16; ++e) Ax.h[e] = (_Float16)0.0f;
    }
#pragma unroll
    for (int e = 0; e < 8; ++e) Ah.h[e] = (_Float16)lds_h[wave][m][hi * 8 + e];
#pragma unroll
    for (int e = 8; e < 16; ++e) Ah.h[e] = (_Float16)0.0f;

    // ---- gates = bias + Xt*W_ih^T + H*W_hh^T via chained WMMA ----
    v8f gate[4];
#pragma unroll
    for (int nt = 0; nt < 4; ++nt) {
      v8f acc;
#pragma unroll
      for (int v = 0; v < 8; ++v) acc[v] = biasv[nt];
      acc = __builtin_amdgcn_wmma_f32_16x16x32_f16(false, Ax.v, false, Bih[nt].v,
                                                   (short)0, acc, false, false);
      acc = __builtin_amdgcn_wmma_f32_16x16x32_f16(false, Ah.v, false, Bhh[nt].v,
                                                   (short)0, acc, false, false);
      gate[nt] = acc;
    }

    // ---- elementwise LSTM cell update in D-layout (branch-free, v_tanh_f32) ----
    v8f hreg;
#pragma unroll
    for (int v = 0; v < 8; ++v) {
      const float ig = fast_sigmoid(gate[0][v]);
      const float fg = fast_sigmoid(gate[1][v]);
      const float gg = fast_tanh(gate[2][v]);
      const float og = fast_sigmoid(gate[3][v]);
      const float c  = fg * cst[v] + ig * gg;
      cst[v]  = c;
      hreg[v] = og * fast_tanh(c);
    }

    // ---- publish h for next timestep's A_h (D-layout -> LDS transpose) ----
    __syncthreads();
#pragma unroll
    for (int v = 0; v < 8; ++v) lds_h[wave][v + hi * 8][m] = hreg[v];
    __syncthreads();
  }

  // ---- final FC: out[g][o] = b_fc[o] + sum_n W_fc[o][n] * h[g][n] ----
#pragma unroll
  for (int rep = 0; rep < 2; ++rep) {
    const int idx = lane + 32 * rep;   // 0..63 -> 16 graphs x 4 outputs
    const int mm  = idx & 15;
    const int oo  = idx >> 4;
    float acc = b_fc[oo];
#pragma unroll
    for (int n = 0; n < T_HIDDEN; ++n) acc += W_fc[oo * T_HIDDEN + n] * lds_h[wave][mm][n];
    out[(gbase + mm) * 4 + oo] = acc;
  }
}

extern "C" void kernel_launch(void* const* d_in, const int* in_sizes, int n_in,
                              void* d_out, int out_size, void* d_ws, size_t ws_size,
                              hipStream_t stream) {
  const float* x     = (const float*)d_in[0];
  const float* W_ih  = (const float*)d_in[1];
  const float* W_hh  = (const float*)d_in[2];
  const float* b_ih  = (const float*)d_in[3];
  const float* b_hh  = (const float*)d_in[4];
  const float* W_fc  = (const float*)d_in[5];
  const float* b_fc  = (const float*)d_in[6];
  const int* edge_index = (const int*)d_in[7];
  const int* edge_attr  = (const int*)d_in[8];
  const int* batch      = (const int*)d_in[9];

  const int E  = in_sizes[8];       // number of edges
  const int ng = out_size / 4;      // number of graphs (4096)
  int* first = (int*)d_ws;          // (ng + 1) ints of scratch

  const int nthreads = ng + 1;
  first_edge_kernel<<<(nthreads + 255) / 256, 256, 0, stream>>>(batch, E, ng, first);

  lstm_fused_kernel<<<ng / 128, 256, 0, stream>>>(
      x, W_ih, W_hh, b_ih, b_hh, W_fc, b_fc,
      edge_index, edge_attr, first, (float*)d_out, E);
}
